// Encoder_Attentioner_3161095930112
// MI455X (gfx1250) — matrix operand order repaired and checked by reading, not compiled, not run on hardware
//
#include <hip/hip_runtime.h>
#include <hip/hip_bf16.h>
#include <math.h>

typedef __attribute__((ext_vector_type(16))) _Float16 v16h;
typedef __attribute__((ext_vector_type(8)))  float    v8f;

// ---------------------------------------------------------------------------
// Stage A1: per-(batch, 64-row chunk) column partial sums of x.  Coalesced.
// grid = 8*16 blocks, 512 threads.
__global__ __launch_bounds__(512)
void colsum_partial(const float* __restrict__ X, float* __restrict__ partial)
{
    const int b     = blockIdx.x >> 4;
    const int chunk = blockIdx.x & 15;
    const int t     = threadIdx.x;
    const float* xp = X + ((size_t)(b * 1024 + chunk * 64)) * 512 + t;
    float acc = 0.f;
    #pragma unroll 8
    for (int r = 0; r < 64; ++r) acc += xp[(size_t)r * 512];
    partial[(b * 16 + chunk) * 512 + t] = acc;
}

// Stage A2: reduce 16 partials -> xsum[8][512].  grid = 8, 512 threads.
__global__ __launch_bounds__(512)
void colsum_finish(const float* __restrict__ partial, float* __restrict__ xsum)
{
    const int b = blockIdx.x, t = threadIdx.x;
    float acc = 0.f;
    #pragma unroll
    for (int j = 0; j < 16; ++j) acc += partial[(b * 16 + j) * 512 + t];
    xsum[b * 512 + t] = acc;
}

// ---------------------------------------------------------------------------
// WMMA GEMM:  out[8][512] = A[8][512] x W(512x512, optionally transposed)
//             + bscale*bias + addSrc.   M padded 8->16 with zeros.
// One wave per 16-column tile; K looped 16 chunks of 32, f16 in / f32 acc.
// grid = 8 blocks x 128 threads = 32 waves (covers N = 512).
__global__ __launch_bounds__(128)
void gemm8x512_wmma(const float* __restrict__ A,
                    const float* __restrict__ W,
                    const float* __restrict__ bias, float bscale,
                    const float* __restrict__ addSrc,
                    float* __restrict__ out,
                    int transposed)
{
    const int lane  = threadIdx.x & 31;
    const int gw    = blockIdx.x * (blockDim.x >> 5) + (threadIdx.x >> 5); // 0..31
    const int nbase = gw * 16;
    const int Ml    = lane & 15;
    const int n     = nbase + Ml;
    const bool hi   = lane >= 16;

    v8f acc = {};
    for (int kk = 0; kk < 16; ++kk) {
        // A fragment: lanes 0-15 -> K = kk*32 + 0..15; lanes 16-31 -> +16.
        const int ka = kk * 32 + (hi ? 16 : 0);
        v16h af;
        if (Ml < 8) {
            const float* ap = A + Ml * 512 + ka;
            #pragma unroll
            for (int e = 0; e < 8; ++e) {
                af[e]     = (_Float16)ap[e];
                af[8 + e] = (_Float16)ap[8 + e];
            }
        } else {
            #pragma unroll
            for (int e = 0; e < 16; ++e) af[e] = (_Float16)0.f;
        }
        // B fragment: lane = column n; lanes 0-15 K = kk*32+0..15, lanes 16-31 +16.
        const int kb = kk * 32 + (hi ? 16 : 0);
        v16h bf;
        if (transposed) {           // Bmat[k][n] = W[n][k]  (contract over W's 2nd dim)
            const float* wp = W + (size_t)n * 512 + kb;
            #pragma unroll
            for (int e = 0; e < 16; ++e) bf[e] = (_Float16)wp[e];
        } else {                    // Bmat[k][n] = W[k][n]  (contract over W's 1st dim)
            const float* wp = W + (size_t)kb * 512 + n;
            #pragma unroll
            for (int e = 0; e < 16; ++e) bf[e] = (_Float16)wp[(size_t)e * 512];
        }
        acc = __builtin_amdgcn_wmma_f32_16x16x32_f16(false, af, false, bf,
                                                     (short)0, acc, false, false);
    }

    if (lane < 16) {            // D: VGPR r = row M=r (batch), lane = column
        #pragma unroll
        for (int r = 0; r < 8; ++r) {
            float v = acc[r];
            if (bias)   v += bscale * bias[n];
            if (addSrc) v += addSrc[r * 512 + n];
            out[r * 512 + n] = v;
        }
    }
}

// v[b] = (sum_b' ksum[b']) - ksum[b].  grid = 8, 512 threads.
__global__ __launch_bounds__(512)
void vcompute(const float* __restrict__ ksum, float* __restrict__ v)
{
    const int b = blockIdx.x, o = threadIdx.x;
    float s = 0.f;
    #pragma unroll
    for (int j = 0; j < 8; ++j) s += ksum[j * 512 + o];
    v[b * 512 + o] = s - ksum[b * 512 + o];
}

// ---------------------------------------------------------------------------
// Main streaming pass: d[row] = x[row] . u[row/1024].  Wave per row,
// coalesced float4 loads, shuffle reduction.  grid = 1024 x 256.
__global__ __launch_bounds__(256)
void rowdot_kernel(const float* __restrict__ X, const float* __restrict__ U,
                   float* __restrict__ d)
{
    const int wid  = threadIdx.x >> 5;
    const int lane = threadIdx.x & 31;
    const int row  = blockIdx.x * 8 + wid;          // 0..8191
    const int b    = row >> 10;
    const float4* xp = (const float4*)(X + (size_t)row * 512);
    const float4* up = (const float4*)(U + (size_t)b * 512);
    float acc = 0.f;
    #pragma unroll
    for (int j = 0; j < 4; ++j) {
        float4 xv = xp[j * 32 + lane];
        float4 uv = up[j * 32 + lane];
        acc += xv.x * uv.x + xv.y * uv.y + xv.z * uv.z + xv.w * uv.w;
    }
    #pragma unroll
    for (int off = 16; off > 0; off >>= 1) acc += __shfl_xor(acc, off, 32);
    if (lane == 0) d[row] = acc;
}

// Per-batch min-max normalize + sigmoid.  grid = 8, 1024 threads.
__global__ __launch_bounds__(1024)
void norm_kernel(const float* __restrict__ d, float* __restrict__ out)
{
    __shared__ float smn[32], smx[32];
    const int b = blockIdx.x, t = threadIdx.x;
    const float v = d[b * 1024 + t];
    float mn = v, mx = v;
    #pragma unroll
    for (int off = 16; off > 0; off >>= 1) {
        mn = fminf(mn, __shfl_xor(mn, off, 32));
        mx = fmaxf(mx, __shfl_xor(mx, off, 32));
    }
    const int wid = t >> 5, lane = t & 31;
    if (lane == 0) { smn[wid] = mn; smx[wid] = mx; }
    __syncthreads();
    if (wid == 0) {
        mn = smn[lane]; mx = smx[lane];
        #pragma unroll
        for (int off = 16; off > 0; off >>= 1) {
            mn = fminf(mn, __shfl_xor(mn, off, 32));
            mx = fmaxf(mx, __shfl_xor(mx, off, 32));
        }
        if (lane == 0) { smn[0] = mn; smx[0] = mx; }
    }
    __syncthreads();
    mn = smn[0]; mx = smx[0];
    const float rep = (v - mn) / (mx - mn);
    const float z   = (rep - 0.65f) / 0.15f;
    out[b * 1024 + t] = 1.f / (1.f + expf(-z));
}

// ---------------------------------------------------------------------------
extern "C" void kernel_launch(void* const* d_in, const int* in_sizes, int n_in,
                              void* d_out, int out_size, void* d_ws, size_t ws_size,
                              hipStream_t stream)
{
    (void)in_sizes; (void)n_in; (void)out_size; (void)ws_size;
    const float* x      = (const float*)d_in[0];   // (8,32,32,512)
    const float* conv_w = (const float*)d_in[1];   // (512,512)
    const float* conv_b = (const float*)d_in[2];   // (512)
    const float* q_w    = (const float*)d_in[3];
    // d_in[4] = q_b: drops out under per-batch min-max normalization.
    const float* k_w    = (const float*)d_in[5];
    const float* k_b    = (const float*)d_in[6];
    float* out = (float*)d_out;

    float* w       = (float*)d_ws;
    float* partial = w;                 // 8*16*512 = 65536
    float* xsum    = w + 65536;         // 4096
    float* xr      = w + 69632;         // 4096
    float* ksum    = w + 73728;         // 4096
    float* vv      = w + 77824;         // 4096
    float* ww      = w + 81920;         // 4096
    float* uu      = w + 86016;         // 4096
    float* dd      = w + 90112;         // 8192  (total 384 KB)

    colsum_partial<<<128, 512, 0, stream>>>(x, partial);
    colsum_finish <<<8,   512, 0, stream>>>(partial, xsum);

    // xr = xsum@conv_w^T + 1024*conv_b + xsum
    gemm8x512_wmma<<<8, 128, 0, stream>>>(xsum, conv_w, conv_b, 1024.f, xsum, xr, 1);
    // ksum = xr@k_w^T + 1024*k_b
    gemm8x512_wmma<<<8, 128, 0, stream>>>(xr, k_w, k_b, 1024.f, nullptr, ksum, 1);
    // v[b] = sum_b' ksum - ksum[b]
    vcompute      <<<8, 512, 0, stream>>>(ksum, vv);
    // w = q_w^T @ v   (contract over q_w's first dim)
    gemm8x512_wmma<<<8, 128, 0, stream>>>(vv, q_w, nullptr, 0.f, nullptr, ww, 0);
    // u = conv_w^T @ w + w
    gemm8x512_wmma<<<8, 128, 0, stream>>>(ww, conv_w, nullptr, 0.f, ww, uu, 0);

    rowdot_kernel<<<1024, 256, 0, stream>>>(x, uu, dd);
    norm_kernel  <<<8,   1024, 0, stream>>>(dd, out);
}
